// Attention_8297876816646
// MI455X (gfx1250) — matrix operand order repaired and checked by reading, not compiled, not run on hardware
//
#include <hip/hip_runtime.h>
#include <hip/hip_bf16.h>

typedef __bf16 bf16_t;
typedef __attribute__((ext_vector_type(16))) __bf16 v16bf;
typedef __attribute__((ext_vector_type(8)))  float  v8f;
typedef unsigned short ushort_t;

union Frag16 { v16bf v; uint4 q[2]; ushort_t h[16]; };
union Acc8   { v8f   v; float f[8]; };

static __device__ __forceinline__ ushort_t f2bf(float f) {
    union { float f; unsigned u; } x; x.f = f;
    unsigned r = x.u + 0x7FFFu + ((x.u >> 16) & 1u);   // round-to-nearest-even
    return (ushort_t)(r >> 16);
}

static __device__ __forceinline__ v8f wmma_bf16(v16bf a, v16bf b, v8f c) {
    return __builtin_amdgcn_wmma_f32_16x16x32_bf16(false, a, false, b, (short)0, c, false, false);
}

// ---------------------------------------------------------------- prep kernels

__global__ void cvt_f32_to_bf16(const float* __restrict__ in, ushort_t* __restrict__ out, int n) {
    int i = blockIdx.x * blockDim.x + threadIdx.x;
    if (i < n) out[i] = f2bf(in[i]);
}

// in: [K][N] f32 row-major  ->  out: [N][K] bf16 row-major
__global__ void transpose_to_bf16(const float* __restrict__ in, ushort_t* __restrict__ out,
                                  int K, int N) {
    int i = blockIdx.x * blockDim.x + threadIdx.x;
    if (i < K * N) {
        int n = i / K, k = i - n * K;
        out[n * K + k] = f2bf(in[k * N + n]);
    }
}

// vT[h][d][s] = qkv_bf16[s][1536 + h*64 + d]   (V transposed per head)
__global__ void v_transpose(const ushort_t* __restrict__ qkvb, ushort_t* __restrict__ vT) {
    int i = blockIdx.x * blockDim.x + threadIdx.x;   // 12*64*4096
    int s = i & 4095;
    int r = i >> 12;
    int d = r & 63;
    int h = r >> 6;
    vT[i] = qkvb[s * 2304 + 1536 + h * 64 + d];
}

// ---------------------------------------------------------------- tiled WMMA GEMM
// C[M][N] = A[M][K] (bf16, row-major) * B[K][N] + bias, with B given transposed
// as Bt[N][K] bf16 row-major. 256 threads, block tile 128x128, K-step 32.
// OUTF32=0 -> bf16 out, OUTF32=1 -> f32 out.

template <int OUTF32>
__global__ __launch_bounds__(256)
void gemm_wmma_bf16(const ushort_t* __restrict__ A, const ushort_t* __restrict__ Bt,
                    const float* __restrict__ bias, void* __restrict__ Cout,
                    int M, int N, int K) {
    __shared__ ushort_t lsA[128 * 32];
    __shared__ ushort_t lsB[128 * 32];

    const int tid  = threadIdx.x;
    const int lane = tid & 31;
    const int wave = tid >> 5;            // 0..7
    const int wM   = wave >> 2;           // 0..1  -> 64 rows each
    const int wN   = wave & 3;            // 0..3  -> 32 cols each
    const int l15  = lane & 15;
    const int hi   = lane >> 4;

    const int mBase = blockIdx.y * 128;
    const int nBase = blockIdx.x * 128;

    Acc8 acc[4][2];
#pragma unroll
    for (int m = 0; m < 4; ++m)
#pragma unroll
        for (int n = 0; n < 2; ++n)
#pragma unroll
            for (int r = 0; r < 8; ++r) acc[m][n].f[r] = 0.0f;

    const int ksteps = K >> 5;
    for (int kt = 0; kt < ksteps; ++kt) {
        // cooperative stage of A (128x32) and Bt (128x32) tiles
#pragma unroll
        for (int it = 0; it < 2; ++it) {
            int slot = tid + it * 256;           // 0..511
            int row  = slot >> 2;
            int c8   = (slot & 3) * 8;
            *(uint4*)&lsA[row * 32 + c8] =
                *(const uint4*)(A + (mBase + row) * K + kt * 32 + c8);
            *(uint4*)&lsB[row * 32 + c8] =
                *(const uint4*)(Bt + (nBase + row) * K + kt * 32 + c8);
        }
        __syncthreads();

        Frag16 af[4], bfr[2];
#pragma unroll
        for (int m = 0; m < 4; ++m) {
            int row = wM * 64 + m * 16 + l15;
            af[m].q[0] = *(const uint4*)&lsA[row * 32 + hi * 8];
            af[m].q[1] = *(const uint4*)&lsA[row * 32 + 16 + hi * 8];
        }
#pragma unroll
        for (int n = 0; n < 2; ++n) {
            int col = wN * 32 + n * 16 + l15;
            bfr[n].q[0] = *(const uint4*)&lsB[col * 32 + hi * 8];
            bfr[n].q[1] = *(const uint4*)&lsB[col * 32 + 16 + hi * 8];
        }
#pragma unroll
        for (int m = 0; m < 4; ++m)
#pragma unroll
            for (int n = 0; n < 2; ++n)
                acc[m][n].v = wmma_bf16(af[m].v, bfr[n].v, acc[m][n].v);
        __syncthreads();
    }

    // epilogue: acc element (vgpr r, lane) -> row = hi*8+r, col = l15
#pragma unroll
    for (int m = 0; m < 4; ++m)
#pragma unroll
        for (int n = 0; n < 2; ++n) {
            int col_g = nBase + wN * 32 + n * 16 + l15;
            float b = bias[col_g];
#pragma unroll
            for (int r = 0; r < 8; ++r) {
                int row_g = mBase + wM * 64 + m * 16 + hi * 8 + r;
                float v = acc[m][n].f[r] + b;
                if (OUTF32)
                    ((float*)Cout)[row_g * N + col_g] = v;
                else
                    ((ushort_t*)Cout)[row_g * N + col_g] = f2bf(v);
            }
        }
}

// ---------------------------------------------------------------- flash attention
// One wave per (16-query strip, head). qkvb: [4096][2304] bf16 (Q|K|V),
// vT: [12][64][4096] bf16, aout: [4096][768] bf16.

__global__ __launch_bounds__(32)
void attn_wmma(const ushort_t* __restrict__ qkvb, const ushort_t* __restrict__ vT,
               ushort_t* __restrict__ aout) {
    __shared__ ushort_t pls[16 * 32];

    const int lane = threadIdx.x;
    const int l15  = lane & 15;
    const int hi   = lane >> 4;
    const int qb   = blockIdx.x * 16;
    const int h    = blockIdx.y;

    // Q A-fragments (16 x 64 = two 16x32 frags), layout per ISA 16-bit A
    Frag16 qf[2];
    const ushort_t* qrow = qkvb + (qb + l15) * 2304 + h * 64;
#pragma unroll
    for (int t = 0; t < 2; ++t) {
        qf[t].q[0] = *(const uint4*)(qrow + t * 32 + hi * 8);
        qf[t].q[1] = *(const uint4*)(qrow + t * 32 + 16 + hi * 8);
    }

    Acc8 o[4];
#pragma unroll
    for (int n = 0; n < 4; ++n)
#pragma unroll
        for (int r = 0; r < 8; ++r) o[n].f[r] = 0.0f;

    float mrow[8], lrow[8];
#pragma unroll
    for (int r = 0; r < 8; ++r) { mrow[r] = -1e30f; lrow[r] = 0.0f; }

    const int nkb = (qb + 47) >> 5;   // key blocks of 32 covering keys 0..qb+15
    for (int kb = 0; kb < nkb; ++kb) {
        const int k0 = kb * 32;

        // S = Q * K^T  (two 16x16 key subtiles)
        Acc8 s[2];
#pragma unroll
        for (int nn = 0; nn < 2; ++nn) {
#pragma unroll
            for (int r = 0; r < 8; ++r) s[nn].f[r] = 0.0f;
            const ushort_t* krow = qkvb + (k0 + nn * 16 + l15) * 2304 + 768 + h * 64;
#pragma unroll
            for (int t = 0; t < 2; ++t) {
                Frag16 kf;
                kf.q[0] = *(const uint4*)(krow + t * 32 + hi * 8);
                kf.q[1] = *(const uint4*)(krow + t * 32 + 16 + hi * 8);
                s[nn].v = wmma_bf16(qf[t].v, kf.v, s[nn].v);
            }
        }

        // scale + causal mask (value-level; EXEC stays all-ones)
#pragma unroll
        for (int nn = 0; nn < 2; ++nn)
#pragma unroll
            for (int r = 0; r < 8; ++r) {
                int kcol = k0 + nn * 16 + l15;
                int qrow_abs = qb + hi * 8 + r;
                float v = s[nn].f[r] * 0.125f;
                s[nn].f[r] = (kcol > qrow_abs) ? -1e30f : v;
            }

        // online softmax: row stats replicated across each 16-lane half
        float fac[8];
#pragma unroll
        for (int r = 0; r < 8; ++r) {
            float mx = fmaxf(s[0].f[r], s[1].f[r]);
#pragma unroll
            for (int msk = 1; msk <= 8; msk <<= 1) mx = fmaxf(mx, __shfl_xor(mx, msk, 32));
            float mn = fmaxf(mrow[r], mx);
            float p0 = __expf(s[0].f[r] - mn);
            float p1 = __expf(s[1].f[r] - mn);
            s[0].f[r] = p0; s[1].f[r] = p1;
            float rs = p0 + p1;
#pragma unroll
            for (int msk = 1; msk <= 8; msk <<= 1) rs += __shfl_xor(rs, msk, 32);
            fac[r]  = __expf(mrow[r] - mn);
            lrow[r] = lrow[r] * fac[r] + rs;
            mrow[r] = mn;
        }
#pragma unroll
        for (int n = 0; n < 4; ++n)
#pragma unroll
            for (int r = 0; r < 8; ++r) o[n].f[r] *= fac[r];

        // P (C-layout) -> LDS -> re-read as A-fragment layout
#pragma unroll
        for (int nn = 0; nn < 2; ++nn)
#pragma unroll
            for (int r = 0; r < 8; ++r)
                pls[(hi * 8 + r) * 32 + nn * 16 + l15] = f2bf(s[nn].f[r]);
        __syncthreads();

        Frag16 pf;
        pf.q[0] = *(const uint4*)&pls[l15 * 32 + hi * 8];
        pf.q[1] = *(const uint4*)&pls[l15 * 32 + 16 + hi * 8];

        // O += P * V  (V^T rows contiguous per lane from vT)
#pragma unroll
        for (int n = 0; n < 4; ++n) {
            Frag16 vf;
            const ushort_t* vrow = vT + (h * 64 + n * 16 + l15) * 4096 + k0 + hi * 8;
            vf.q[0] = *(const uint4*)(vrow);
            vf.q[1] = *(const uint4*)(vrow + 16);
            o[n].v = wmma_bf16(pf.v, vf.v, o[n].v);
        }
        __syncthreads();
    }

    // normalize and emit merged-head output (bf16 for the proj GEMM)
#pragma unroll
    for (int n = 0; n < 4; ++n)
#pragma unroll
        for (int r = 0; r < 8; ++r) {
            float v = o[n].f[r] / lrow[r];
            aout[(qb + hi * 8 + r) * 768 + h * 64 + n * 16 + l15] = f2bf(v);
        }
}

// ---------------------------------------------------------------- launcher

extern "C" void kernel_launch(void* const* d_in, const int* in_sizes, int n_in,
                              void* d_out, int out_size, void* d_ws, size_t ws_size,
                              hipStream_t stream) {
    (void)in_sizes; (void)n_in; (void)out_size; (void)ws_size;
    const float* x      = (const float*)d_in[0];   // [4096][768]
    const float* w_attn = (const float*)d_in[1];   // [768][2304]
    const float* b_attn = (const float*)d_in[2];   // [2304]
    const float* w_proj = (const float*)d_in[3];   // [768][768]
    const float* b_proj = (const float*)d_in[4];   // [768]
    float* out = (float*)d_out;                    // [4096][768]

    const int S = 4096, D = 768, TD = 2304, H = 12;

    ushort_t* ws    = (ushort_t*)d_ws;
    ushort_t* xb     = ws;                       // S*D
    ushort_t* wattnT = xb     + S * D;           // TD*D
    ushort_t* wprojT = wattnT + TD * D;          // D*D
    ushort_t* qkvb   = wprojT + D * D;           // S*TD
    ushort_t* vT     = qkvb   + S * TD;          // H*64*S
    ushort_t* aout   = vT     + H * 64 * S;      // S*D

    int nx = S * D;
    cvt_f32_to_bf16<<<(nx + 255) / 256, 256, 0, stream>>>(x, xb, nx);

    int na = D * TD;
    transpose_to_bf16<<<(na + 255) / 256, 256, 0, stream>>>(w_attn, wattnT, D, TD);
    int np = D * D;
    transpose_to_bf16<<<(np + 255) / 256, 256, 0, stream>>>(w_proj, wprojT, D, D);

    gemm_wmma_bf16<0><<<dim3(TD / 128, S / 128), 256, 0, stream>>>(
        xb, wattnT, b_attn, (void*)qkvb, S, TD, D);

    int nv = H * 64 * S;
    v_transpose<<<(nv + 255) / 256, 256, 0, stream>>>(qkvb, vT);

    attn_wmma<<<dim3(S / 16, H), 32, 0, stream>>>(qkvb, vT, aout);

    gemm_wmma_bf16<1><<<dim3(D / 128, S / 128), 256, 0, stream>>>(
        aout, wprojT, b_proj, (void*)out, S, D, D);
}
